// PhotoVoModel_24206435680890
// MI455X (gfx1250) — matrix operand order repaired and checked by reading, not compiled, not run on hardware
//
#include <hip/hip_runtime.h>

typedef __attribute__((ext_vector_type(16))) _Float16 v16h;
typedef __attribute__((ext_vector_type(8)))  _Float16 v8h;
typedef __attribute__((ext_vector_type(8)))  float    v8f;

#define BB   8
#define NN   4096
#define MM   1024
#define PP   16
#define DD   384
#define HH   640
#define WW   640
#define RROWS 16384      /* B * 2M */
#define PDD   768        /* 3*P*P  */

__device__ __forceinline__ int imin(int a, int b) { return a < b ? a : b; }
__device__ __forceinline__ int imax(int a, int b) { return a > b ? a : b; }

// ---------------------------------------------------------------------------
// 1) Match gather + patch extraction -> fp16 A matrix (16384 x 768)
// ---------------------------------------------------------------------------
__global__ __launch_bounds__(256)
void patch_kernel(const float* __restrict__ im0, const float* __restrict__ im1,
                  const float* __restrict__ kp0, const float* __restrict__ kp1,
                  const float* __restrict__ sc0, const float* __restrict__ sc1,
                  const int*   __restrict__ sm,
                  _Float16* __restrict__ Ap, float* __restrict__ coords,
                  float* __restrict__ scout)
{
    int r   = blockIdx.x;          // 0..16383  == b*2M + (img*M + m)
    int b   = r >> 11;             // / 2048
    int t2  = r & 2047;
    int img = t2 >> 10;            // 0 or 1
    int m   = t2 & 1023;

    int idx0  = sm[(b * MM + m) * 2 + 0];
    int idx1  = sm[(b * MM + m) * 2 + 1];
    int vflag = idx1 > -1;
    int idx   = img ? idx1 : idx0;
    int i     = imin(imax(idx, 0), NN - 1);

    const float* kp = img ? kp1 : kp0;
    const float* sc = img ? sc1 : sc0;
    const float* im = img ? im1 : im0;

    float kx = vflag ? kp[((size_t)b * NN + i) * 2 + 0] : -1.0f;
    float ky = vflag ? kp[((size_t)b * NN + i) * 2 + 1] : -1.0f;
    float sv = vflag ? sc[(size_t)b * NN + i] : 0.0f;

    int x0 = imin(imax((int)rintf(kx) - PP / 2, 0), WW - PP);
    int y0 = imin(imax((int)rintf(ky) - PP / 2, 0), HH - PP);

    int tid = threadIdx.x;         // 256 threads -> one 16x16 plane position
    int py = tid >> 4, px = tid & 15;
#pragma unroll
    for (int c = 0; c < 3; ++c) {
        float v = im[(((size_t)b * 3 + c) * HH + (y0 + py)) * WW + (x0 + px)];
        v = fminf(fmaxf(v, 0.0f), 1.0f);
        Ap[(size_t)r * PDD + c * 256 + tid] = (_Float16)v;
    }
    if (tid == 0) {
        coords[(size_t)r * 2 + 0] = kx;
        coords[(size_t)r * 2 + 1] = ky;
        scout[r] = sv;
    }
}

// ---------------------------------------------------------------------------
// 2) f32 weight (K x Nsrc slice) -> fp16 transposed (N x K) for WMMA B tiles
// ---------------------------------------------------------------------------
__global__ void cvt_wT_kernel(const float* __restrict__ src, _Float16* __restrict__ dst,
                              int K, int Nn, int stride, int coloff)
{
    int idx = blockIdx.x * 256 + threadIdx.x;
    if (idx >= K * Nn) return;
    int n = idx / K;
    int k = idx - n * K;
    dst[idx] = (_Float16)src[(size_t)k * stride + coloff + n];
}

// ---------------------------------------------------------------------------
// 3) pos MLP hidden:  H[r][j] = relu(cx*w1[0][j] + cy*w1[1][j] + b1[j])  (fp16)
// ---------------------------------------------------------------------------
__global__ void hidden_kernel(const float* __restrict__ coords,
                              const float* __restrict__ w1, const float* __restrict__ b1,
                              _Float16* __restrict__ Hout)
{
    int idx = blockIdx.x * 256 + threadIdx.x;
    if (idx >= RROWS * DD) return;
    int r = idx / DD;
    int j = idx - r * DD;
    float cx = coords[(size_t)r * 2 + 0];
    float cy = coords[(size_t)r * 2 + 1];
    float v  = fmaf(cx, w1[j], fmaf(cy, w1[DD + j], b1[j]));
    Hout[idx] = (_Float16)fmaxf(v, 0.0f);
}

// ---------------------------------------------------------------------------
// 4) Tiled WMMA GEMM:  C[M x 384] = A[M x K](f16) * Bt[384 x K]^T (+bias)
//    Block tile 128x64, 8 waves of 32x32, K-step 32, double-buffered LDS.
//    Epilogue: optional Cin accumulate, rank-1 (rowscale*colvec), relu,
//              f32 and/or f16 outputs.
// ---------------------------------------------------------------------------
#define LDA 40   /* halves; 80B row -> 16B aligned segments, bank-skewed */
#define LDB 40

__global__ __launch_bounds__(256)
void wmma_gemm_kernel(const _Float16* __restrict__ A, const _Float16* __restrict__ Bt,
                      const float* __restrict__ bias, const float* __restrict__ Cin,
                      const float* __restrict__ rowscale, const float* __restrict__ colvec,
                      float* __restrict__ C32, _Float16* __restrict__ C16,
                      int Ktot, int relu)
{
    __shared__ _Float16 As[2][128 * LDA];
    __shared__ _Float16 Bs[2][64 * LDB];

    const int blockM = blockIdx.x * 128;
    const int blockN = blockIdx.y * 64;
    const int tid  = threadIdx.x;
    const int wv   = tid >> 5;
    const int lane = tid & 31;
    const int half = lane >> 4;     // 0: lanes 0-15, 1: lanes 16-31
    const int l16  = lane & 15;
    const int m0w  = (wv >> 1) * 32;
    const int n0w  = (wv & 1) * 32;

    // per-thread staging coordinates
    const int arow = tid >> 1;
    const int aseg = (tid & 1) * 16;
    const int brow = tid >> 2;
    const int bseg = (tid & 3) * 8;
    const _Float16* agp = A  + (size_t)(blockM + arow) * Ktot + aseg;
    const _Float16* bgp = Bt + (size_t)(blockN + brow) * Ktot + bseg;

    auto stageA = [&](int buf, int k0) {
        const v8h* src = (const v8h*)(agp + k0);
        v8h* dst = (v8h*)&As[buf][arow * LDA + aseg];
        dst[0] = src[0];
        dst[1] = src[1];
    };
    auto stageB = [&](int buf, int k0) {
        const v8h* src = (const v8h*)(bgp + k0);
        v8h* dst = (v8h*)&Bs[buf][brow * LDB + bseg];
        dst[0] = src[0];
    };

    v8f zero = {};
    v8f acc[2][2];
#pragma unroll
    for (int i = 0; i < 2; ++i)
#pragma unroll
        for (int j = 0; j < 2; ++j) {
            if (Cin) {
                union { v8f v; float f[8]; } u;
#pragma unroll
                for (int r = 0; r < 8; ++r) {
                    int row = blockM + m0w + i * 16 + r + half * 8;
                    int col = blockN + n0w + j * 16 + l16;
                    u.f[r] = Cin[(size_t)row * DD + col];
                }
                acc[i][j] = u.v;
            } else {
                acc[i][j] = zero;
            }
        }

    const int nsteps = Ktot >> 5;
    stageA(0, 0);
    stageB(0, 0);
    __syncthreads();

    for (int s = 0; s < nsteps; ++s) {
        const int cur = s & 1;
        const int nxt = cur ^ 1;

        // fragment loads from current buffer (issued first; the staging
        // ds_stores below can stay outstanding past the WMMAs)
        union FragU { v16h v; v8h h[2]; };
        FragU fa[2], fb[2];
        // A fragment (16x32 f16): lanes 0-15 hold K 0-7/16-23, lanes 16-31 K 8-15/24-31
#pragma unroll
        for (int i = 0; i < 2; ++i) {
            const v8h* p = (const v8h*)&As[cur][(m0w + i * 16 + l16) * LDA + half * 8];
            fa[i].h[0] = p[0];   // K: half*8    .. half*8+7
            fa[i].h[1] = p[2];   // K: 16+half*8 .. 16+half*8+7
        }
        // B fragment (32x16 f16): lanes 0-15 hold K 0-7/16-23, lanes 16-31 K 8-15/24-31
#pragma unroll
        for (int j = 0; j < 2; ++j) {
            const v8h* q = (const v8h*)&Bs[cur][(n0w + j * 16 + l16) * LDB + half * 8];
            fb[j].h[0] = q[0];
            fb[j].h[1] = q[2];
        }

        // stage next K-tile into the other buffer (overlaps with WMMA below)
        if (s + 1 < nsteps) {
            int k0n = (s + 1) << 5;
            stageA(nxt, k0n);
            stageB(nxt, k0n);
        }
        // prefetch K+2 tiles into cache (global_prefetch_b8)
        if (s + 2 < nsteps) {
            int k0p = (s + 2) << 5;
            __builtin_prefetch((const void*)(agp + k0p), 0, 3);
            __builtin_prefetch((const void*)(bgp + k0p), 0, 3);
        }

#pragma unroll
        for (int i = 0; i < 2; ++i)
#pragma unroll
            for (int j = 0; j < 2; ++j)
                acc[i][j] = __builtin_amdgcn_wmma_f32_16x16x32_f16(
                    false, fa[i].v, false, fb[j].v, (short)0, acc[i][j], false, false);

        __syncthreads();
    }

    // epilogue
#pragma unroll
    for (int i = 0; i < 2; ++i)
#pragma unroll
        for (int j = 0; j < 2; ++j) {
            union { v8f v; float f[8]; } u; u.v = acc[i][j];
            int col = blockN + n0w + j * 16 + l16;
            float bv = bias ? bias[col] : 0.0f;
            float cv = rowscale ? colvec[col] : 0.0f;
#pragma unroll
            for (int r = 0; r < 8; ++r) {
                int row = blockM + m0w + i * 16 + r + half * 8;
                float v = u.f[r] + bv;
                if (rowscale) v += rowscale[row] * cv;
                if (relu) v = fmaxf(v, 0.0f);
                if (C32) C32[(size_t)row * DD + col] = v;
                if (C16) C16[(size_t)row * DD + col] = (_Float16)v;
            }
        }
}

// ---------------------------------------------------------------------------
// 5) Row LayerNorm over 384 (biased var, eps 1e-5); f32 and/or f16 out
// ---------------------------------------------------------------------------
__global__ __launch_bounds__(128)
void ln_kernel(const float* __restrict__ in, const float* __restrict__ g,
               const float* __restrict__ bta, float* __restrict__ out32,
               _Float16* __restrict__ out16)
{
    int row = blockIdx.x;
    int tid = threadIdx.x;
    __shared__ float s1[128], s2[128];
    __shared__ float mu_s, rs_s;
    const float* ip = in + (size_t)row * DD;
    float x0 = ip[tid], x1 = ip[tid + 128], x2 = ip[tid + 256];
    s1[tid] = x0 + x1 + x2;
    s2[tid] = x0 * x0 + x1 * x1 + x2 * x2;
    __syncthreads();
    for (int off = 64; off; off >>= 1) {
        if (tid < off) { s1[tid] += s1[tid + off]; s2[tid] += s2[tid + off]; }
        __syncthreads();
    }
    if (tid == 0) {
        float mu  = s1[0] * (1.0f / DD);
        float var = s2[0] * (1.0f / DD) - mu * mu;
        mu_s = mu;
        rs_s = rsqrtf(var + 1e-5f);
    }
    __syncthreads();
    float mu = mu_s, rs = rs_s;
#pragma unroll
    for (int p = 0; p < 3; ++p) {
        int j = tid + p * 128;
        float x = (p == 0) ? x0 : (p == 1) ? x1 : x2;
        float y = (x - mu) * rs * g[j] + bta[j];
        if (out32) out32[(size_t)row * DD + j] = y;
        if (out16) out16[(size_t)row * DD + j] = (_Float16)y;
    }
}

// ---------------------------------------------------------------------------
// 6) q = image_embs @ Wq + bq   (8 x 384, tiny)
// ---------------------------------------------------------------------------
__global__ void q_kernel(const float* __restrict__ emb, const float* __restrict__ win,
                         const float* __restrict__ bin, float* __restrict__ qbuf)
{
    int b = blockIdx.x, n = threadIdx.x;   // 384 threads
    float acc = bin[n];
    for (int k = 0; k < DD; ++k)
        acc += emb[(size_t)b * DD + k] * win[(size_t)k * (3 * DD) + n];
    qbuf[(size_t)b * DD + n] = acc;
}

// ---------------------------------------------------------------------------
// 7) single-query attention per (batch, head): softmax over 2048 keys
// ---------------------------------------------------------------------------
__global__ __launch_bounds__(256)
void attn_kernel(const float* __restrict__ qbuf, const float* __restrict__ Kbuf,
                 const float* __restrict__ Vbuf, float* __restrict__ obuf)
{
    int b = blockIdx.x >> 3;
    int h = blockIdx.x & 7;
    int tid = threadIdx.x;
    __shared__ float qs[48];
    __shared__ float lg[2048];
    __shared__ float red[256];
    __shared__ float oacc[4][48];
    if (tid < 48) qs[tid] = qbuf[(size_t)b * DD + h * 48 + tid];
    __syncthreads();

    float lmax = -3.0e38f;
    for (int n = tid; n < 2048; n += 256) {
        const float* kp = Kbuf + ((size_t)b * 2048 + n) * DD + h * 48;
        float s = 0.0f;
#pragma unroll
        for (int d = 0; d < 48; ++d) s += qs[d] * kp[d];
        s *= 0.14433756729740643f;   // 1/sqrt(48)
        lg[n] = s;
        lmax = fmaxf(lmax, s);
    }
    red[tid] = lmax; __syncthreads();
    for (int off = 128; off; off >>= 1) {
        if (tid < off) red[tid] = fmaxf(red[tid], red[tid + off]);
        __syncthreads();
    }
    float m = red[0]; __syncthreads();

    float se = 0.0f;
    for (int n = tid; n < 2048; n += 256) {
        float e = expf(lg[n] - m);
        lg[n] = e;
        se += e;
    }
    red[tid] = se; __syncthreads();
    for (int off = 128; off; off >>= 1) {
        if (tid < off) red[tid] += red[tid + off];
        __syncthreads();
    }
    float inv = 1.0f / red[0];

    // weighted V sum: 4 key-chunks x 48 dims (192 active lanes), LDS reduce
    int chunk = tid >> 6;          // 0..3
    int d     = tid & 63;
    float acc = 0.0f;
    if (d < 48) {
        const float* vp = Vbuf + ((size_t)b * 2048 + chunk * 512) * DD + h * 48 + d;
        for (int n = 0; n < 512; ++n)
            acc += lg[chunk * 512 + n] * vp[(size_t)n * DD];
        oacc[chunk][d] = acc;
    }
    __syncthreads();
    if (tid < 48) {
        float o = oacc[0][tid] + oacc[1][tid] + oacc[2][tid] + oacc[3][tid];
        obuf[(size_t)b * DD + h * 48 + tid] = o * inv;
    }
}

// ---------------------------------------------------------------------------
// 8) head: (o @ attn_out_w + b) @ fc_w + fc_b -> out (8 x 6)
// ---------------------------------------------------------------------------
__global__ void head_kernel(const float* __restrict__ obuf, const float* __restrict__ Wout,
                            const float* __restrict__ bout, const float* __restrict__ fc_w,
                            const float* __restrict__ fc_b, float* __restrict__ out)
{
    int b = blockIdx.x, tid = threadIdx.x;   // 384 threads
    __shared__ float tmp[DD];
    float acc = bout[tid];
    for (int k = 0; k < DD; ++k)
        acc += obuf[(size_t)b * DD + k] * Wout[(size_t)k * DD + tid];
    tmp[tid] = acc;
    __syncthreads();
    if (tid < 6) {
        float r = fc_b[tid];
        for (int k = 0; k < DD; ++k) r += tmp[k] * fc_w[k * 6 + tid];
        out[b * 6 + tid] = r;
    }
}

// ---------------------------------------------------------------------------
extern "C" void kernel_launch(void* const* d_in, const int* in_sizes, int n_in,
                              void* d_out, int out_size, void* d_ws, size_t ws_size,
                              hipStream_t stream)
{
    (void)in_sizes; (void)n_in; (void)out_size; (void)ws_size;
    const float* image0     = (const float*)d_in[0];
    const float* image1     = (const float*)d_in[1];
    const float* kpts0      = (const float*)d_in[2];
    const float* kpts1      = (const float*)d_in[3];
    const float* scores0    = (const float*)d_in[4];
    const float* scores1    = (const float*)d_in[5];
    const int*   smatches   = (const int*)  d_in[6];
    const float* image_embs = (const float*)d_in[7];
    const float* pp_w       = (const float*)d_in[8];
    const float* pp_b       = (const float*)d_in[9];
    const float* ln1_g      = (const float*)d_in[10];
    const float* ln1_b      = (const float*)d_in[11];
    const float* pos_w1     = (const float*)d_in[12];
    const float* pos_b1     = (const float*)d_in[13];
    const float* pos_w2     = (const float*)d_in[14];
    const float* pos_b2     = (const float*)d_in[15];
    const float* sc_w       = (const float*)d_in[16];
    const float* sc_b       = (const float*)d_in[17];
    const float* ln2_g      = (const float*)d_in[18];
    const float* ln2_b      = (const float*)d_in[19];
    const float* attn_in_w  = (const float*)d_in[20];
    const float* attn_in_b  = (const float*)d_in[21];
    const float* attn_out_w = (const float*)d_in[22];
    const float* attn_out_b = (const float*)d_in[23];
    const float* fc_w       = (const float*)d_in[24];
    const float* fc_b       = (const float*)d_in[25];

    char* ws = (char*)d_ws;
    size_t off = 0;
    auto alloc = [&](size_t bytes) -> void* {
        void* p = ws + off;
        off = (off + bytes + 255) & ~(size_t)255;
        return p;
    };

    _Float16* Ap      = (_Float16*)alloc((size_t)RROWS * PDD * 2);
    float*    coords  = (float*)   alloc((size_t)RROWS * 2 * 4);
    float*    scoresv = (float*)   alloc((size_t)RROWS * 4);
    float*    buf1    = (float*)   alloc((size_t)RROWS * DD * 4);  // pf_pre / pre-LN2 / Vbuf
    float*    buf2    = (float*)   alloc((size_t)RROWS * DD * 4);  // pf(LN1) / Kbuf
    _Float16* H16     = (_Float16*)alloc((size_t)RROWS * DD * 2);
    _Float16* X2      = (_Float16*)alloc((size_t)RROWS * DD * 2);
    _Float16* PE      = (_Float16*)alloc((size_t)RROWS * DD * 2);
    float*    qbuf    = (float*)   alloc((size_t)BB * DD * 4);
    float*    obuf    = (float*)   alloc((size_t)BB * DD * 4);
    _Float16* wT_pp   = (_Float16*)alloc((size_t)DD * PDD * 2);
    _Float16* wT_pos2 = (_Float16*)alloc((size_t)DD * DD * 2);
    _Float16* wT_sc   = (_Float16*)alloc((size_t)DD * DD * 2);
    _Float16* wT_k    = (_Float16*)alloc((size_t)DD * DD * 2);
    _Float16* wT_v    = (_Float16*)alloc((size_t)DD * DD * 2);

    // 1) patches + coords + scores
    patch_kernel<<<RROWS, 256, 0, stream>>>(image0, image1, kpts0, kpts1,
                                            scores0, scores1, smatches,
                                            Ap, coords, scoresv);
    // 2) fp16 transposed weights
    cvt_wT_kernel<<<(PDD * DD + 255) / 256, 256, 0, stream>>>(pp_w,      wT_pp,   PDD, DD, DD,   0);
    cvt_wT_kernel<<<(DD * DD + 255) / 256, 256, 0, stream>>>(pos_w2,    wT_pos2, DD,  DD, DD,   0);
    cvt_wT_kernel<<<(DD * DD + 255) / 256, 256, 0, stream>>>(sc_w,      wT_sc,   DD,  DD, DD,   0);
    cvt_wT_kernel<<<(DD * DD + 255) / 256, 256, 0, stream>>>(attn_in_w, wT_k,    DD,  DD, 3*DD, DD);
    cvt_wT_kernel<<<(DD * DD + 255) / 256, 256, 0, stream>>>(attn_in_w, wT_v,    DD,  DD, 3*DD, 2*DD);
    // 3) pos MLP hidden (fp16)
    hidden_kernel<<<(RROWS * DD + 255) / 256, 256, 0, stream>>>(coords, pos_w1, pos_b1, H16);

    dim3 ggrid(RROWS / 128, DD / 64);
    // 4) GEMM1: pf_pre = patches @ pp_w + pp_b            -> buf1 (f32)
    wmma_gemm_kernel<<<ggrid, 256, 0, stream>>>(Ap, wT_pp, pp_b,
                                                nullptr, nullptr, nullptr,
                                                buf1, nullptr, PDD, 0);
    // 5) LN1 -> buf2 (f32)
    ln_kernel<<<RROWS, 128, 0, stream>>>(buf1, ln1_g, ln1_b, buf2, nullptr);
    // 6) GEMM2: X2 = LN1 + H @ pos_w2 + pos_b2            -> X2 (f16)
    wmma_gemm_kernel<<<ggrid, 256, 0, stream>>>(H16, wT_pos2, pos_b2,
                                                buf2, nullptr, nullptr,
                                                nullptr, X2, DD, 0);
    // 7) GEMM3: relu(X2 @ sc_w[:384] + score*sc_w[384] + sc_b) -> buf1 (f32)
    wmma_gemm_kernel<<<ggrid, 256, 0, stream>>>(X2, wT_sc, sc_b,
                                                nullptr, scoresv, sc_w + (size_t)DD * DD,
                                                buf1, nullptr, DD, 1);
    // 8) LN2 -> PE (f16)
    ln_kernel<<<RROWS, 128, 0, stream>>>(buf1, ln2_g, ln2_b, nullptr, PE);
    // 9) GEMM4: K = PE @ Wk + bk                          -> buf2 (f32)
    wmma_gemm_kernel<<<ggrid, 256, 0, stream>>>(PE, wT_k, attn_in_b + DD,
                                                nullptr, nullptr, nullptr,
                                                buf2, nullptr, DD, 0);
    // 10) GEMM5: V = PE @ Wv + bv                         -> buf1 (f32)
    wmma_gemm_kernel<<<ggrid, 256, 0, stream>>>(PE, wT_v, attn_in_b + 2 * DD,
                                                nullptr, nullptr, nullptr,
                                                buf1, nullptr, DD, 0);
    // 11) q, attention, head
    q_kernel<<<BB, DD, 0, stream>>>(image_embs, attn_in_w, attn_in_b, qbuf);
    attn_kernel<<<BB * 8, 256, 0, stream>>>(qbuf, buf2, buf1, obuf);
    head_kernel<<<BB, DD, 0, stream>>>(obuf, attn_out_w, attn_out_b, fc_w, fc_b,
                                       (float*)d_out);
}
